// liGRU_52759378264453
// MI455X (gfx1250) — matrix operand order repaired and checked by reading, not compiled, not run on hardware
//
#include <hip/hip_runtime.h>
#include <cstdint>
#include <cstddef>

// ---------------------------------------------------------------------------
// Li-GRU (2 layers, T=500 B=64 I=512 H=1024) for MI455X / gfx1250.
//   1) cvt f32->bf16 for x and the 8 weight matrices
//   2) proj_gemm: az_pre = Xb @ Wxz^T + b, ah_pre = Xb @ Wxh^T + b   (WMMA bf16,
//      double-buffered LDS X tiles filled with global_load_async_to_lds_b128)
//   3) bn: training-mode BatchNorm over the batch axis per (t,h)
//   4) scan: persistent 64-WG kernel; each WG owns 16 output columns, keeps its
//      W_hz/W_hh slice in VGPR B-fragments for all 500 steps, ping-pongs h
//      through L2 as bf16, one device barrier per timestep.
// ---------------------------------------------------------------------------

typedef unsigned short ushort_t;
typedef __attribute__((ext_vector_type(16))) __bf16 v16bf;
typedef __attribute__((ext_vector_type(8)))  float  v8f;

union FragBF { v16bf v; uint4 q[2]; };

__device__ inline ushort_t f2bf(float f) {
  unsigned u = __float_as_uint(f);
  u += 0x7FFFu + ((u >> 16) & 1u);        // round-to-nearest-even
  return (ushort_t)(u >> 16);
}

__device__ inline v8f zero_v8f() {
  v8f z;
  for (int i = 0; i < 8; ++i) z[i] = 0.f;
  return z;
}

__device__ inline v8f wmma_bf16(v16bf a, v16bf b, v8f c) {
#if defined(__gfx1250__)
  // (neg_a, A, neg_b, B, c_mod, C, reuse_a, reuse_b)
  return __builtin_amdgcn_wmma_f32_16x16x32_bf16(false, a, false, b, (short)0, c,
                                                 false, false);
#else
  return c;
#endif
}

// ---- async global->LDS copy (16B per lane) --------------------------------
__device__ inline void async_copy_b128(void* lds, const void* gp) {
#if defined(__gfx1250__)
  // LDS aperture: low 32 bits of a generic shared pointer are the LDS offset.
  unsigned loff = (unsigned)(uintptr_t)lds;
  asm volatile("global_load_async_to_lds_b128 %0, %1, off"
               :: "v"(loff), "v"(gp) : "memory");
#else
  *(uint4*)lds = *(const uint4*)gp;
#endif
}

__device__ inline void async_wait_2() {
#if defined(__gfx1250__)
  asm volatile("s_wait_asynccnt 2" ::: "memory");
#endif
}
__device__ inline void async_wait_0() {
#if defined(__gfx1250__)
  asm volatile("s_wait_asynccnt 0" ::: "memory");
#endif
}

// ---------------------------------------------------------------------------
// f32 -> bf16 convert
// ---------------------------------------------------------------------------
__global__ __launch_bounds__(256) void cvt_bf16_kernel(const float* __restrict__ s,
                                                       ushort_t* __restrict__ d,
                                                       long n) {
  long i = (long)blockIdx.x * 256 + threadIdx.x;
  long stride = (long)gridDim.x * 256;
  for (; i < n; i += stride) d[i] = f2bf(s[i]);
}

// ---------------------------------------------------------------------------
// Zero h ping/pong buffers + grid barrier
// ---------------------------------------------------------------------------
__global__ __launch_bounds__(256) void scan_init_kernel(ushort_t* h0, ushort_t* h1,
                                                        int n, int* bar) {
  int i = blockIdx.x * 256 + threadIdx.x;
  if (i < n) { h0[i] = 0; h1[i] = 0; }
  if (i == 0) { bar[0] = 0; bar[1] = 0; }
}

// ---------------------------------------------------------------------------
// Projection GEMM: Cz/Ch[M,1024] = Xb[M,K] @ W{z,h}[1024,K]^T + bias
// block tile 128x64, K-step 32, 8 waves (4x2), each wave 32x32 (2x2 wmma tiles)
// ---------------------------------------------------------------------------
#define PBM 128
#define PBN 64
#define PBK 32

__global__ __launch_bounds__(256) void proj_gemm_kernel(
    const ushort_t* __restrict__ X, int K,
    const ushort_t* __restrict__ Wz, const ushort_t* __restrict__ Wh,
    const float* __restrict__ bz, const float* __restrict__ bh,
    float* __restrict__ Cz, float* __restrict__ Ch) {
  __shared__ ushort_t xt[2][PBM * PBK];           // 2 x 8 KB, double buffered

  const int tid  = threadIdx.x;
  const int lane = tid & 31;
  const int wave = tid >> 5;
  const int wm = wave & 3;                        // row group (32 rows each)
  const int wn = wave >> 2;                       // col group (32 cols each)
  const int bm = blockIdx.x * PBM;
  const int bn = blockIdx.y * PBN;
  const int lo = lane & 15;
  const int hi = lane >> 4;

  auto stage = [&](int buf, int koff) {
    // 512 chunks of 16B; each thread copies 2
    for (int c = tid; c < (PBM * PBK) / 8; c += 256) {
      int row = c >> 2, q = c & 3;
      const ushort_t* src = X + (size_t)(bm + row) * K + koff + q * 8;
      async_copy_b128(&xt[buf][row * PBK + q * 8], src);
    }
  };

  v8f accz[2][2], acch[2][2];
  for (int mi = 0; mi < 2; ++mi)
    for (int ni = 0; ni < 2; ++ni) { accz[mi][ni] = zero_v8f(); acch[mi][ni] = zero_v8f(); }

  stage(0, 0);
  const int nk = K / PBK;
  for (int kt = 0; kt < nk; ++kt) {
    const int cur = kt & 1;
    if (kt + 1 < nk) { stage(cur ^ 1, (kt + 1) * PBK); async_wait_2(); }
    else             { async_wait_0(); }
    __syncthreads();

    // B fragments straight from global (L2-resident weights), contiguous k order, same as A:
    // lane lo = N, K run of 16 starting at hi*16 -> one contiguous 32B read.
    FragBF bzf[2], bhf[2];
#pragma unroll
    for (int ni = 0; ni < 2; ++ni) {
      int col = bn + wn * 32 + ni * 16 + lo;
      size_t wo = (size_t)col * K + (size_t)kt * PBK + hi * 16;
      const uint4* pz = (const uint4*)(Wz + wo);
      bzf[ni].q[0] = pz[0]; bzf[ni].q[1] = pz[1];
      const uint4* ph = (const uint4*)(Wh + wo);
      bhf[ni].q[0] = ph[0]; bhf[ni].q[1] = ph[1];
    }
#pragma unroll
    for (int mi = 0; mi < 2; ++mi) {
      // A layout: lane row = lo, K run of 16 starting at hi*16
      int row = wm * 32 + mi * 16 + lo;
      FragBF a;
      a.q[0] = *(const uint4*)&xt[cur][row * PBK + hi * 16];
      a.q[1] = *(const uint4*)&xt[cur][row * PBK + hi * 16 + 8];
#pragma unroll
      for (int ni = 0; ni < 2; ++ni) {
        accz[mi][ni] = wmma_bf16(a.v, bzf[ni].v, accz[mi][ni]);
        acch[mi][ni] = wmma_bf16(a.v, bhf[ni].v, acch[mi][ni]);
      }
    }
    __syncthreads();
  }

  // epilogue: C/D layout row = r + 8*hi, col = lo
#pragma unroll
  for (int mi = 0; mi < 2; ++mi)
#pragma unroll
    for (int ni = 0; ni < 2; ++ni) {
      int gcol = bn + wn * 32 + ni * 16 + lo;
      float biasz = bz[gcol], biash = bh[gcol];
#pragma unroll
      for (int r = 0; r < 8; ++r) {
        int grow = bm + wm * 32 + mi * 16 + r + hi * 8;
        size_t o = (size_t)grow * 1024 + gcol;
        Cz[o] = accz[mi][ni][r] + biasz;
        Ch[o] = acch[mi][ni][r] + biash;
      }
    }
}

// ---------------------------------------------------------------------------
// BatchNorm over batch axis (B=64) per (t,h), biased variance, in place.
// ---------------------------------------------------------------------------
__global__ __launch_bounds__(256) void bn_kernel(float* __restrict__ a,
                                                 const float* __restrict__ gamma,
                                                 const float* __restrict__ beta,
                                                 int TH, int Bn, int Hn) {
  int idx = blockIdx.x * 256 + threadIdx.x;
  if (idx >= TH) return;
  int t = idx / Hn, h = idx - t * Hn;
  float* base = a + (size_t)t * Bn * Hn + h;
  float s = 0.f, s2 = 0.f;
  for (int b = 0; b < Bn; ++b) { float v = base[(size_t)b * Hn]; s += v; s2 += v * v; }
  float invn = 1.f / (float)Bn;
  float mu = s * invn;
  float var = s2 * invn - mu * mu;
  float inv = rsqrtf(var + 1e-5f);
  float g = gamma[h], be = beta[h];
  for (int b = 0; b < Bn; ++b) {
    size_t o = (size_t)b * Hn;
    base[o] = g * (base[o] - mu) * inv + be;
  }
}

// ---------------------------------------------------------------------------
// Recurrent scan. 64 persistent WGs, 16 columns each.
// Waves 0..3: z-gate GEMM, waves 4..7: h-candidate GEMM; wave kg owns
// k-steps {kg, kg+4, ..., kg+28} with its 8 weight B-fragments in VGPRs.
// Deterministic cross-wave reduction through LDS. One grid barrier per step.
// ---------------------------------------------------------------------------
#define SCAN_WGS 64
#define NCOL 16

__global__ __launch_bounds__(256) void scan_kernel(
    const ushort_t* __restrict__ Whz, const ushort_t* __restrict__ Whh,
    const float* __restrict__ az, const float* __restrict__ ah,
    const float* __restrict__ bhz, const float* __restrict__ bhh,
    ushort_t* __restrict__ hA, ushort_t* __restrict__ hB,
    ushort_t* __restrict__ ybf, float* __restrict__ yf32,
    int T, int* __restrict__ bar) {
  __shared__ float red[8][64 * NCOL];             // 32 KB partials
  __shared__ float hslice[64 * NCOL];             // this WG's f32 h slice

  const int tid  = threadIdx.x;
  const int lane = tid & 31;
  const int wave = tid >> 5;
  const int lo = lane & 15;
  const int hi = lane >> 4;
  const int colbase = blockIdx.x * NCOL;
  const int kg = wave & 3;
  const ushort_t* W = (wave >> 2) ? Whh : Whz;

  // Load this wave's weight B-fragments once; resident for all 500 steps.
  FragBF Bf[8];
#pragma unroll
  for (int i = 0; i < 8; ++i) {
    int ks = kg + i * 4;                          // k-step index (K = ks*32)
    size_t wo = (size_t)(colbase + lo) * 1024 + (size_t)ks * 32 + hi * 16;
    const uint4* p = (const uint4*)(W + wo);
    Bf[i].q[0] = p[0]; Bf[i].q[1] = p[1];
  }
  for (int i = tid; i < 64 * NCOL; i += 256) hslice[i] = 0.f;
  __syncthreads();

  for (int t = 0; t < T; ++t) {
    const ushort_t* hin = (t & 1) ? hB : hA;
    ushort_t* hout      = (t & 1) ? hA : hB;

    v8f acc[4];
#pragma unroll
    for (int mt = 0; mt < 4; ++mt) acc[mt] = zero_v8f();

#pragma unroll
    for (int mt = 0; mt < 4; ++mt) {
#pragma unroll
      for (int i = 0; i < 8; ++i) {
        int koff = (kg + i * 4) * 32;
        const ushort_t* hp = hin + (size_t)(mt * 16 + lo) * 1024 + koff;
        FragBF a;
        a.q[0] = *(const uint4*)(hp + hi * 16);
        a.q[1] = *(const uint4*)(hp + hi * 16 + 8);
        acc[mt] = wmma_bf16(a.v, Bf[i].v, acc[mt]);
      }
    }

    // deposit partials (fixed slot per wave -> deterministic reduction)
#pragma unroll
    for (int mt = 0; mt < 4; ++mt)
#pragma unroll
      for (int r = 0; r < 8; ++r)
        red[wave][(mt * 16 + r + hi * 8) * NCOL + lo] = acc[mt][r];
    __syncthreads();

    // gate epilogue: 1024 (row,col) cells, 4 per thread, fixed-order sums
    for (int i = tid; i < 64 * NCOL; i += 256) {
      int row = i >> 4, c = i & 15;
      int gcol = colbase + c;
      float sz = (red[0][i] + red[1][i]) + (red[2][i] + red[3][i]);
      float sh = (red[4][i] + red[5][i]) + (red[6][i] + red[7][i]);
      size_t o = ((size_t)t * 64 + row) * 1024 + gcol;
      float azv = az[o] + sz + bhz[gcol];
      float ahv = ah[o] + sh + bhh[gcol];
      float Z  = 1.f / (1.f + expf(-azv));
      float Ht = ahv > 0.f ? ahv : 0.f;
      float h2 = Z * hslice[i] + (1.f - Z) * Ht;
      hslice[i] = h2;
      hout[(size_t)row * 1024 + gcol] = f2bf(h2);
      if (ybf)  ybf[o] = f2bf(h2);
      if (yf32) yf32[o] = h2;
    }

    // device-wide barrier (sense-reversal via generation counter)
    __threadfence();
    __syncthreads();
    if (tid == 0) {
      int gen  = __hip_atomic_load(bar + 1, __ATOMIC_RELAXED, __HIP_MEMORY_SCOPE_AGENT);
      int prev = __hip_atomic_fetch_add(bar, 1, __ATOMIC_ACQ_REL, __HIP_MEMORY_SCOPE_AGENT);
      if (prev == SCAN_WGS - 1) {
        __hip_atomic_store(bar, 0, __ATOMIC_RELAXED, __HIP_MEMORY_SCOPE_AGENT);
        __hip_atomic_fetch_add(bar + 1, 1, __ATOMIC_RELEASE, __HIP_MEMORY_SCOPE_AGENT);
      } else {
        while (__hip_atomic_load(bar + 1, __ATOMIC_ACQUIRE, __HIP_MEMORY_SCOPE_AGENT) == gen) {
#if defined(__gfx1250__)
          __builtin_amdgcn_s_sleep(2);
#endif
        }
      }
    }
    __syncthreads();
  }
}

// ---------------------------------------------------------------------------
// Host launcher
// ---------------------------------------------------------------------------
extern "C" void kernel_launch(void* const* d_in, const int* in_sizes, int n_in,
                              void* d_out, int out_size, void* d_ws, size_t ws_size,
                              hipStream_t stream) {
  (void)in_sizes; (void)n_in; (void)out_size; (void)ws_size;
  const int T = 500, B = 64, I = 512, H = 1024;
  const long nX   = (long)T * B * I;       // 16,384,000
  const long nTBH = (long)T * B * H;       // 32,768,000
  const long HI = (long)H * I, HH = (long)H * H;

  auto inf = [&](int i) { return (const float*)d_in[i]; };
  // input order: x, then per layer: W_xz,b_xz,g_z,beta_z,W_hz,b_hz,W_xh,b_xh,g_h,beta_h,W_hh,b_hh

  char* ws = (char*)d_ws;
  size_t off = 0;
  auto alloc = [&](size_t bytes) -> char* {
    char* p = ws + off;
    off = (off + bytes + 255) & ~(size_t)255;
    return p;
  };
  ushort_t* xb    = (ushort_t*)alloc((size_t)nX * 2);
  ushort_t* wxz0b = (ushort_t*)alloc((size_t)HI * 2);
  ushort_t* wxh0b = (ushort_t*)alloc((size_t)HI * 2);
  ushort_t* whz0b = (ushort_t*)alloc((size_t)HH * 2);
  ushort_t* whh0b = (ushort_t*)alloc((size_t)HH * 2);
  ushort_t* wxz1b = (ushort_t*)alloc((size_t)HH * 2);
  ushort_t* wxh1b = (ushort_t*)alloc((size_t)HH * 2);
  ushort_t* whz1b = (ushort_t*)alloc((size_t)HH * 2);
  ushort_t* whh1b = (ushort_t*)alloc((size_t)HH * 2);
  float*    azb   = (float*)alloc((size_t)nTBH * 4);
  float*    ahb   = (float*)alloc((size_t)nTBH * 4);
  ushort_t* y0b   = (ushort_t*)alloc((size_t)nTBH * 2);
  ushort_t* hping = (ushort_t*)alloc((size_t)B * H * 2);
  ushort_t* hpong = (ushort_t*)alloc((size_t)B * H * 2);
  int*      bar   = (int*)alloc(256);

  // 1) converts
  cvt_bf16_kernel<<<4096, 256, 0, stream>>>((const float*)d_in[0], xb, nX);
  cvt_bf16_kernel<<<1024, 256, 0, stream>>>(inf(1),  wxz0b, HI);
  cvt_bf16_kernel<<<1024, 256, 0, stream>>>(inf(7),  wxh0b, HI);
  cvt_bf16_kernel<<<2048, 256, 0, stream>>>(inf(5),  whz0b, HH);
  cvt_bf16_kernel<<<2048, 256, 0, stream>>>(inf(11), whh0b, HH);
  cvt_bf16_kernel<<<2048, 256, 0, stream>>>(inf(13), wxz1b, HH);
  cvt_bf16_kernel<<<2048, 256, 0, stream>>>(inf(19), wxh1b, HH);
  cvt_bf16_kernel<<<2048, 256, 0, stream>>>(inf(17), whz1b, HH);
  cvt_bf16_kernel<<<2048, 256, 0, stream>>>(inf(23), whh1b, HH);

  const int M = T * B;                           // 32000
  dim3 pgrid(M / PBM, H / PBN);                  // (250, 16)
  const int TH = T * H, bnBlocks = (TH + 255) / 256;

  // ---- layer 0 ----
  proj_gemm_kernel<<<pgrid, 256, 0, stream>>>(xb, I, wxz0b, wxh0b, inf(2), inf(8), azb, ahb);
  bn_kernel<<<bnBlocks, 256, 0, stream>>>(azb, inf(3),  inf(4),  TH, B, H);
  bn_kernel<<<bnBlocks, 256, 0, stream>>>(ahb, inf(9),  inf(10), TH, B, H);
  scan_init_kernel<<<(B * H + 255) / 256, 256, 0, stream>>>(hping, hpong, B * H, bar);
  scan_kernel<<<SCAN_WGS, 256, 0, stream>>>(whz0b, whh0b, azb, ahb, inf(6), inf(12),
                                            hping, hpong, y0b, nullptr, T, bar);

  // ---- layer 1 ----
  proj_gemm_kernel<<<pgrid, 256, 0, stream>>>(y0b, H, wxz1b, wxh1b, inf(14), inf(20), azb, ahb);
  bn_kernel<<<bnBlocks, 256, 0, stream>>>(azb, inf(15), inf(16), TH, B, H);
  bn_kernel<<<bnBlocks, 256, 0, stream>>>(ahb, inf(21), inf(22), TH, B, H);
  scan_init_kernel<<<(B * H + 255) / 256, 256, 0, stream>>>(hping, hpong, B * H, bar);
  scan_kernel<<<SCAN_WGS, 256, 0, stream>>>(whz1b, whh1b, azb, ahb, inf(18), inf(24),
                                            hping, hpong, nullptr, (float*)d_out, T, bar);
}
